// LeadingZeroDetector16_29875792511517
// MI455X (gfx1250) — matrix operand order repaired and checked by reading, not compiled, not run on hardware
//
#include <hip/hip_runtime.h>

typedef __attribute__((ext_vector_type(16))) _Float16     v16h;
typedef __attribute__((ext_vector_type(8)))  float        v8f;
typedef __attribute__((ext_vector_type(8)))  int          v8i;
typedef __attribute__((ext_vector_type(4)))  unsigned int u4v;

// LZC16: per row of 16 {0,1} floats, emit the 4-bit binary (MSB first) of the
// index of the first set bit; all-zero rows emit [1,1,1,1].
//   lzc = is_first · POS + all_zero · [1,1,1,1]
// computed as one v_wmma_f32_16x16x32_f16 per 16 rows:
//   A[16x32] f16 : one-hot is_first encoded as 2^-14 (0x0400); K=16 = all_zero
//   B[32x16] f16 : POS bits (and row 16 = ones) scaled by 2^14 (0x7400)
// so every nonzero product is exactly 1.0 in the f32 accumulator.
__global__ __launch_bounds__(256) void lzc16_wmma_kernel(const float* __restrict__ X,
                                                         float* __restrict__ out,
                                                         int n_tiles) {
    const int lane = (int)(threadIdx.x & 31);
    const int col  = lane & 15;          // A: row-in-tile; B/D: column
    const int half = lane >> 4;          // 0: bits 0..7, 1: bits 8..15
    const int nWaves = (int)((gridDim.x * blockDim.x) >> 5);
    int tile = (int)(blockIdx.x * (blockDim.x >> 5)) + (int)(threadIdx.x >> 5);

    // ---- B (32x16 f16) constant, built once per wave ----
    // lane n holds column n&15; lanes 0-15 rows K=0..7,16..23, lanes 16-31 rows K=8..15,24..31.
    unsigned int breg[8];
    const int kbase = half << 3;
    #pragma unroll
    for (int i = 0; i < 8; ++i) {
        const int k0 = kbase + 2 * i + (i >= 4 ? 8 : 0), k1 = k0 + 1;
        unsigned int lo = 0u, hi = 0u;
        if (col < 4) {
            const int sh = (3 - col) & 31;
            lo = (k0 < 16) ? (((unsigned)k0 >> sh) & 1u) * 0x7400u
                           : (k0 == 16 ? 0x7400u : 0u);
            hi = (k1 < 16) ? (((unsigned)k1 >> sh) & 1u) * 0x7400u
                           : (k1 == 16 ? 0x7400u : 0u);
        }
        breg[i] = lo | (hi << 16);
    }
    const v8i bi = {(int)breg[0], (int)breg[1], (int)breg[2], (int)breg[3],
                    (int)breg[4], (int)breg[5], (int)breg[6], (int)breg[7]};
    const v16h b = __builtin_bit_cast(v16h, bi);

    // per-lane constant offsets; pointers are strided per iteration
    const float* pin = X + (size_t)tile * 256 + (size_t)col * 16 + (size_t)(half << 3);
    float* pout = out + (size_t)tile * 64 + (size_t)((half << 3) * 4 + col);
    const size_t instep  = (size_t)nWaves * 256;
    const size_t outstep = (size_t)nWaves * 64;

    while (tile < n_tiles) {
        // ---- load 8 elements (two b128); bit j of m = (x_j == 1.0f) ----
        u4v u0 = *(const u4v*)pin;
        u4v u1 = *(const u4v*)(pin + 4);
        const unsigned int uu[8] = {u0.x, u0.y, u0.z, u0.w, u1.x, u1.y, u1.z, u1.w};
        unsigned int m = 0u;
        #pragma unroll
        for (int j = 0; j < 8; ++j) m |= ((uu[j] >> 29) & 1u) << j;   // 1.0f has bit29

        // ---- cross-half exchange: one ds_bpermute ----
        const unsigned int mOther =
            (unsigned int)__builtin_amdgcn_ds_bpermute((lane ^ 16) << 2, (int)m);

        // first set bit in this chunk, gated off if an earlier chunk had one;
        // p == 8 means "no contribution from this lane"
        const bool earlier = (half != 0) && (mOther != 0u);
        const unsigned int gated = earlier ? 0x100u : (m | 0x100u);
        const unsigned int p = (unsigned int)__builtin_ctz(gated);

        // ---- A (16x32 f16): 0x0400 at element p; K=16 slot = all_zero ----
        const unsigned int v = 0x0400u << ((p & 1u) << 4);
        const unsigned int r = p >> 1;
        const unsigned int a0 = (r == 0u) ? v : 0u;
        const unsigned int a1 = (r == 1u) ? v : 0u;
        const unsigned int a2 = (r == 2u) ? v : 0u;
        const unsigned int a3 = (r == 3u) ? v : 0u;
        const unsigned int az = (((m | mOther) == 0u) && half == 0) ? 0x0400u : 0u;
        const v8i ai = {(int)a0, (int)a1, (int)a2, (int)a3, (int)az, 0, 0, 0};
        const v16h a = __builtin_bit_cast(v16h, ai);

        // ---- D = A*B : one WMMA per 16 rows ----
        v8f c = {};
        c = __builtin_amdgcn_wmma_f32_16x16x32_f16(false, a, false, b, (short)0, c,
                                                   false, false);

        // ---- D layout: c[rr] @ lane n(<4) -> row rr, col n; lane n+16 -> row rr+8 ----
        if (col < 4) {
            #pragma unroll
            for (int rr = 0; rr < 8; ++rr) pout[(size_t)rr * 4] = c[rr];
        }

        tile += nWaves;
        pin  += instep;
        pout += outstep;
    }
}

extern "C" void kernel_launch(void* const* d_in, const int* in_sizes, int n_in,
                              void* d_out, int out_size, void* d_ws, size_t ws_size,
                              hipStream_t stream) {
    const float* X = (const float*)d_in[0];
    float* out = (float*)d_out;

    const int rows    = in_sizes[0] / 16;   // 4194304
    const int n_tiles = rows / 16;          // 262144

    // ~8 tiles per wave to amortize setup; 256 threads = 8 waves per block
    int blocks = n_tiles / 64;
    if (blocks < 1) blocks = 1;

    lzc16_wmma_kernel<<<blocks, 256, 0, stream>>>(X, out, n_tiles);
}
